// ClassNetPP_20229295965040
// MI455X (gfx1250) — matrix operand order repaired and checked by reading, not compiled, not run on hardware
//
#include <hip/hip_runtime.h>
#include <math.h>

typedef __attribute__((ext_vector_type(16))) _Float16 v16h;
typedef __attribute__((ext_vector_type(8)))  _Float16 v8h;
typedef __attribute__((ext_vector_type(8)))  float    v8f;
typedef __attribute__((ext_vector_type(4)))  float    v4f;

#define TAU_F 0.25f
#define BN_RSQ 0.9999950000375f   /* 1/sqrt(1+1e-5) */

// ---------------------------------------------------------------------------
// WMMA fragment conventions (wave32, V_WMMA_F32_16X16X32_F16):
//  A (16x32 f16): lane l holds row M = l&15; with h = l>>4 the v16h elements
//    map to K = (i>>3)*16 + h*8 + (i&7)  -> two contiguous 8-element runs at
//    K offsets h*8 and 16+h*8.
//  B (32x16 f16): lane l holds column N = l&15; elements i map to
//    K = (i>>3)*16 + h*8 + (i&7) (same as A). Weights are PRE-PACKED into this order so each
//    lane issues one contiguous 32B load per (ktile, ntile).
//  C/D (16x16 f32): v8f element r -> row M = r + (l>>4)*8, col N = l&15.
// Activations are stored in f16 between stages: A fragments become two raw
// 16B loads (no cvt), halving L2 traffic of every intermediate stream.
// ---------------------------------------------------------------------------

static __device__ __forceinline__ v16h load_a_frag_f32(const float* __restrict__ arow,
                                                       int kt, int h) {
  const float* p = arow + (kt << 5) + (h << 3);
  v4f x0 = *(const v4f*)(p + 0);
  v4f x1 = *(const v4f*)(p + 4);
  v4f x2 = *(const v4f*)(p + 16);
  v4f x3 = *(const v4f*)(p + 20);
  v16h a;
  a[0]=(_Float16)x0[0];  a[1]=(_Float16)x0[1];  a[2]=(_Float16)x0[2];  a[3]=(_Float16)x0[3];
  a[4]=(_Float16)x1[0];  a[5]=(_Float16)x1[1];  a[6]=(_Float16)x1[2];  a[7]=(_Float16)x1[3];
  a[8]=(_Float16)x2[0];  a[9]=(_Float16)x2[1];  a[10]=(_Float16)x2[2]; a[11]=(_Float16)x2[3];
  a[12]=(_Float16)x3[0]; a[13]=(_Float16)x3[1]; a[14]=(_Float16)x3[2]; a[15]=(_Float16)x3[3];
  return a;
}

static __device__ __forceinline__ v16h load_a_frag_f16(const _Float16* __restrict__ arow,
                                                       int kt, int h) {
  const _Float16* p = arow + (kt << 5) + (h << 3);
  v8h x0 = *(const v8h*)(p);        // K = h*8 .. h*8+7
  v8h x1 = *(const v8h*)(p + 16);   // K = 16+h*8 .. 16+h*8+7
  v16h a;
#pragma unroll
  for (int i = 0; i < 8; ++i) { a[i] = x0[i]; a[8 + i] = x1[i]; }
  return a;
}

// Pack f32 weight matrix w[Cout, Cin] (element stride `estride`, offset `eoff`
// for conv-tap slicing) into f16 B-fragments laid out [kt][ntile][lane][16].
__global__ __launch_bounds__(256) void pack_b_kernel(
    const float* __restrict__ src, _Float16* __restrict__ dst,
    int Cout, int Cin, int estride, int eoff) {
  int t = blockIdx.x * 256 + threadIdx.x;
  if (t >= Cin * Cout) return;
  int e    = t & 15;
  int lane = (t >> 4) & 31;
  int q    = t >> 9;
  int NT   = Cout >> 4;
  int ntl  = q % NT;
  int kt   = q / NT;
  int k = (kt << 5) + ((e >> 3) << 4) + ((lane & 16) ? 8 : 0) + (e & 7);
  int n = (ntl << 4) + (lane & 15);
  dst[t] = (_Float16)src[((size_t)n * Cin + k) * estride + eoff];
}

// Generic WMMA GEMM: out[m,n] = epi( sum_k A[m,k] * W[n,k] ).
// Each wave computes 16 rows x (16*NTW) cols reusing one A fragment NTW times.
// AF16: A stored f16 (raw fragment loads); OF16: store output as f16.
// Epilogue: optional BN (gamma,beta), optional f16 residual, optional ReLU.
template <int NTW, bool AF16, bool OF16>
__global__ __launch_bounds__(128) void wmma_gemm(
    const void* __restrict__ Av, int lda,
    const _Float16* __restrict__ Bp, int K, int N,
    void* __restrict__ outv,
    const float* __restrict__ gamma, const float* __restrict__ beta,
    const _Float16* __restrict__ residual, int relu) {
  int wave = threadIdx.x >> 5, lane = threadIdx.x & 31;
  int h = lane >> 4, cn = lane & 15;
  int m0 = blockIdx.x << 4;
  int NT = N >> 4;
  int ntBase = (blockIdx.y * 4 + wave) * NTW;

  const float*    Af = (const float*)Av    + (AF16 ? 0 : (size_t)(m0 + cn) * lda);
  const _Float16* Ah = (const _Float16*)Av + (AF16 ? (size_t)(m0 + cn) * lda : 0);

  v8f acc[NTW];
#pragma unroll
  for (int j = 0; j < NTW; ++j) { v8f zz = {}; acc[j] = zz; }

  int ktn = K >> 5;
#pragma unroll 2
  for (int kt = 0; kt < ktn; ++kt) {
    v16h a;
    if (AF16) a = load_a_frag_f16(Ah, kt, h);
    else      a = load_a_frag_f32(Af, kt, h);
#pragma unroll
    for (int j = 0; j < NTW; ++j) {
      const _Float16* bp = Bp + (((size_t)kt * NT + ntBase + j) * 32 + lane) * 16;
      v16h b = *(const v16h*)bp;
      acc[j] = __builtin_amdgcn_wmma_f32_16x16x32_f16(
          false, a, false, b, (short)0, acc[j], false, false);
    }
  }

#pragma unroll
  for (int j = 0; j < NTW; ++j) {
    int n = (ntBase + j) * 16 + cn;
    float g = 1.f, bt = 0.f;
    if (gamma) { g = gamma[n] * BN_RSQ; bt = beta[n]; }
#pragma unroll
    for (int r = 0; r < 8; ++r) {
      int m = m0 + r + (h << 3);
      float v = acc[j][r];
      if (gamma)    v = v * g + bt;
      if (residual) v += (float)residual[(size_t)m * N + n];
      if (relu)     v = fmaxf(v, 0.f);
      if (OF16) ((_Float16*)outv)[(size_t)m * N + n] = (_Float16)v;
      else      ((float*)outv)[(size_t)m * N + n] = v;
    }
  }
}

// Dual dilated 3x3 conv (Cin=Cout=128, f16 in/out) as 9 shifted K=128 WMMA
// GEMMs. ntile 0..7 -> local (dil=1) ch 0..127; ntile 8..15 -> global (dil=2)
// ch 128..255 of zcat. One M-tile = 16 consecutive pixels of one image row.
__global__ __launch_bounds__(128) void wmma_conv(
    const _Float16* __restrict__ z, const _Float16* __restrict__ packL,
    const _Float16* __restrict__ packG, _Float16* __restrict__ zcat) {
  int wave = threadIdx.x >> 5, lane = threadIdx.x & 31;
  int h = lane >> 4, cn = lane & 15;
  int nt = blockIdx.y * 4 + wave;           // 0..15
  int conv = nt >> 3, ntl = nt & 7;
  const _Float16* pk = conv ? packG : packL;
  int dil = conv ? 2 : 1;
  int m0 = blockIdx.x << 4;
  int bidx = m0 >> 10, rem = m0 & 1023;
  int yy = rem >> 5, x0 = rem & 31;
  int xr = x0 + cn;                         // this lane's row-pixel x

  v8f acc = {};
  for (int ky = 0; ky < 3; ++ky) {
    int ny = yy + (ky - 1) * dil;
    if ((unsigned)ny >= 32u) continue;      // wave-uniform
    for (int kx = 0; kx < 3; ++kx) {
      int nx = xr + (kx - 1) * dil;
      bool vld = (unsigned)nx < 32u;        // per-lane edge predicate
      int nidx = (bidx << 10) + (ny << 5) + (vld ? nx : 0);
      const _Float16* ar = z + (size_t)nidx * 128;
      int tap = ky * 3 + kx;
#pragma unroll
      for (int kt = 0; kt < 4; ++kt) {
        v16h a = {};
        if (vld) a = load_a_frag_f16(ar, kt, h);
        const _Float16* bp = pk + (((size_t)(tap * 4 + kt) * 8 + ntl) * 32 + lane) * 16;
        v16h b = *(const v16h*)bp;
        acc = __builtin_amdgcn_wmma_f32_16x16x32_f16(
            false, a, false, b, (short)0, acc, false, false);
      }
    }
  }
#pragma unroll
  for (int r = 0; r < 8; ++r) {
    int m = m0 + r + (h << 3);
    int n = nt * 16 + cn;
    zcat[(size_t)m * 256 + n] = (_Float16)fmaxf(acc[r], 0.f);
  }
}

// L2-normalize each 512-wide f16 row -> f32 feats; one wave per row.
__global__ __launch_bounds__(256) void normalize_kernel(
    const _Float16* __restrict__ xf, float* __restrict__ feats) {
  int wave = threadIdx.x >> 5, lane = threadIdx.x & 31;
  int m = blockIdx.x * 8 + wave;
  const _Float16* row = xf + (size_t)m * 512;
  float v[16];
  float ss = 0.f;
#pragma unroll
  for (int i = 0; i < 16; ++i) { v[i] = (float)row[i * 32 + lane]; ss += v[i] * v[i]; }
#pragma unroll
  for (int o = 16; o > 0; o >>= 1) ss += __shfl_xor(ss, o, 32);
  float rn = 1.f / fmaxf(sqrtf(ss), 1e-12f);
  float* orow = feats + (size_t)m * 512;
#pragma unroll
  for (int i = 0; i < 16; ++i) orow[i * 32 + lane] = v[i] * rn;
}

// cam[m,c] = TAU*(LSE_k(sims[m,c*32+k]/TAU) - log 32); lane c<20 per row.
__global__ __launch_bounds__(128) void cam_kernel(
    const _Float16* __restrict__ sims, float* __restrict__ act) {
  int wave = threadIdx.x >> 5, lane = threadIdx.x & 31;
  int m = blockIdx.x * 4 + wave;
  if (lane >= 20) return;
  const _Float16* s = sims + (size_t)m * 640 + lane * 32;
  float sv[32];
#pragma unroll
  for (int i = 0; i < 32; ++i) sv[i] = (float)s[i];
  float mx = -1e30f;
#pragma unroll
  for (int i = 0; i < 32; ++i) mx = fmaxf(mx, sv[i]);
  float sum = 0.f;
#pragma unroll
  for (int i = 0; i < 32; ++i) sum += expf((sv[i] - mx) * (1.f / TAU_F));
  float cam = TAU_F * logf(sum) + mx - TAU_F * logf(32.f);
  int b = m >> 10, n = m & 1023;
  act[((size_t)b * 20 + lane) * 1024 + n] = cam;
}

// logits[bc] = TAU*(LSE_n(cam/TAU) - log 1024) * logit_scale; one wave per (b,c).
__global__ __launch_bounds__(32) void logits_kernel(
    const float* __restrict__ act, const float* __restrict__ lsc,
    float* __restrict__ logits) {
  int lane = threadIdx.x;
  const float* a = act + (size_t)blockIdx.x * 1024;
  float mx = -1e30f, sum = 0.f;
  for (int i = 0; i < 32; ++i) {
    float u = a[i * 32 + lane] * (1.f / TAU_F);
    if (u > mx) { sum = sum * expf(mx - u) + 1.f; mx = u; }
    else        { sum += expf(u - mx); }
  }
#pragma unroll
  for (int o = 16; o > 0; o >>= 1) {
    float mo = __shfl_xor(mx, o, 32);
    float so = __shfl_xor(sum, o, 32);
    float M = fmaxf(mx, mo);
    sum = sum * expf(mx - M) + so * expf(mo - M);
    mx = M;
  }
  if (lane == 0)
    logits[blockIdx.x] = TAU_F * (mx + logf(sum) - logf(1024.f)) * lsc[0];
}

extern "C" void kernel_launch(void* const* d_in, const int* in_sizes, int n_in,
                              void* d_out, int out_size, void* d_ws, size_t ws_size,
                              hipStream_t stream) {
  (void)in_sizes; (void)n_in; (void)out_size; (void)ws_size;
  const float* tokens   = (const float*)d_in[0];
  const float* w_in     = (const float*)d_in[1];
  const float* bn1_g    = (const float*)d_in[2];
  const float* bn1_b    = (const float*)d_in[3];
  const float* w_reduce = (const float*)d_in[4];
  const float* w_local  = (const float*)d_in[5];
  const float* w_global = (const float*)d_in[6];
  const float* w_rcb    = (const float*)d_in[7];
  const float* bn2_g    = (const float*)d_in[8];
  const float* bn2_b    = (const float*)d_in[9];
  const float* w_out    = (const float*)d_in[10];
  const float* protos   = (const float*)d_in[11];
  const float* lscale   = (const float*)d_in[12];

  const int B = 32, Npix = 1024, Din = 768, D = 512, Hd = 128, C = 20, Kp = 32;
  const int M = B * Npix;  // 32768 pixel rows

  // ---- workspace carve (256B aligned) ----
  char* ws = (char*)d_ws;
  size_t off = 0;
  auto take = [&](size_t bytes) -> char* {
    char* p = ws + off;
    off = (off + bytes + 255) & ~(size_t)255;
    return p;
  };
  _Float16* pin   = (_Float16*)take(sizeof(_Float16) * (size_t)Din * D);
  _Float16* pred  = (_Float16*)take(sizeof(_Float16) * (size_t)D * Hd);
  _Float16* ploc  = (_Float16*)take(sizeof(_Float16) * 9 * (size_t)Hd * Hd);
  _Float16* pglo  = (_Float16*)take(sizeof(_Float16) * 9 * (size_t)Hd * Hd);
  _Float16* prcb  = (_Float16*)take(sizeof(_Float16) * (size_t)(2 * Hd) * D);
  _Float16* pout  = (_Float16*)take(sizeof(_Float16) * (size_t)D * D);
  _Float16* pprot = (_Float16*)take(sizeof(_Float16) * (size_t)D * (C * Kp));
  _Float16* y    = (_Float16*)take(sizeof(_Float16) * (size_t)M * D);       // 32 MB
  _Float16* z    = (_Float16*)take(sizeof(_Float16) * (size_t)M * Hd);      // 8 MB
  _Float16* zcat = (_Float16*)take(sizeof(_Float16) * (size_t)M * 2 * Hd);  // 16 MB
  _Float16* xf   = (_Float16*)take(sizeof(_Float16) * (size_t)M * D);       // 32 MB
  _Float16* sims = z;  // reuse dead z/zcat/xf span (56 MB) for sims (40 MB)

  float* d_logits = (float*)d_out;
  float* d_act    = d_logits + B * C;
  float* d_feats  = d_act + (size_t)B * C * Npix;

  // ---- pack all weights to f16 WMMA B-fragments ----
  pack_b_kernel<<<(Din * D) / 256, 256, 0, stream>>>(w_in, pin, D, Din, 1, 0);
  pack_b_kernel<<<(D * Hd) / 256, 256, 0, stream>>>(w_reduce, pred, Hd, D, 1, 0);
  for (int t = 0; t < 9; ++t) {
    pack_b_kernel<<<(Hd * Hd) / 256, 256, 0, stream>>>(
        w_local,  ploc + (size_t)t * Hd * Hd, Hd, Hd, 9, t);
    pack_b_kernel<<<(Hd * Hd) / 256, 256, 0, stream>>>(
        w_global, pglo + (size_t)t * Hd * Hd, Hd, Hd, 9, t);
  }
  pack_b_kernel<<<(2 * Hd * D) / 256, 256, 0, stream>>>(w_rcb, prcb, D, 2 * Hd, 1, 0);
  pack_b_kernel<<<(D * D) / 256, 256, 0, stream>>>(w_out, pout, D, D, 1, 0);
  pack_b_kernel<<<(D * C * Kp) / 256, 256, 0, stream>>>(protos, pprot, C * Kp, D, 1, 0);

  dim3 gblk(128);
  // y = relu(bn1(tokens @ w_in^T))            [A f32 input, out f16]
  wmma_gemm<4, false, true><<<dim3(M / 16, 2), gblk, 0, stream>>>(
      tokens, Din, pin, Din, D, y, bn1_g, bn1_b, nullptr, 1);
  // z = relu(y @ w_reduce^T)                  [f16 -> f16]
  wmma_gemm<2, true, true><<<dim3(M / 16, 1), gblk, 0, stream>>>(
      y, D, pred, D, Hd, z, nullptr, nullptr, nullptr, 1);
  // zcat = relu([conv3x3_d1(z), conv3x3_d2(z)])
  wmma_conv<<<dim3(M / 16, 4), gblk, 0, stream>>>(z, ploc, pglo, zcat);
  // y = relu(bn2(zcat @ w_rcb^T) + y)         [in-place f16 residual]
  wmma_gemm<4, true, true><<<dim3(M / 16, 2), gblk, 0, stream>>>(
      zcat, 2 * Hd, prcb, 2 * Hd, D, y, bn2_g, bn2_b, y, 1);
  // xf = y @ w_out^T                          [f16 -> f16]
  wmma_gemm<4, true, true><<<dim3(M / 16, 2), gblk, 0, stream>>>(
      y, D, pout, D, D, xf, nullptr, nullptr, nullptr, 0);
  // feats = normalize(xf) -> f32 output
  normalize_kernel<<<M / 8, 256, 0, stream>>>(xf, d_feats);
  // sims = feats @ protos^T                   [A f32 (output buf), out f16]
  wmma_gemm<5, false, true><<<dim3(M / 16, 2), gblk, 0, stream>>>(
      d_feats, D, pprot, D, C * Kp, sims, nullptr, nullptr, nullptr, 0);
  // act maps (tau-LSE over K) -> output
  cam_kernel<<<M / 4, 128, 0, stream>>>(sims, d_act);
  // logits (tau-LSE over N, scaled) -> output
  logits_kernel<<<B * C, 32, 0, stream>>>(d_act, lscale, d_logits);
}
